// MaskedAttention_51307679318299
// MI455X (gfx1250) — matrix operand order repaired and checked by reading, not compiled, not run on hardware
//
#include <hip/hip_runtime.h>
#include <hip/hip_bf16.h>
#include <stdint.h>

// ---------------------------------------------------------------------------
// MaskedAttention for MI455X (gfx1250): bf16 WMMA pipeline + async-LDS staging
// shapes: B=2, S=2048, DIM=2048, NH=16, NKV=8, HD=128
// ---------------------------------------------------------------------------

#define BB   2
#define SS   2048
#define DIMM 2048
#define NHH  16
#define NKVV 8
#define HDD  128
#define ROWS (BB * SS)          // 4096
#define SCALE 0.08838834764831845f  // 1/sqrt(128)

typedef __attribute__((ext_vector_type(16))) __bf16    v16bf;
typedef __attribute__((ext_vector_type(8)))  float     v8f;
typedef __attribute__((ext_vector_type(8)))  unsigned int v8u;

#define DEV __device__ __forceinline__

DEV unsigned short f2bf(float f) {
  unsigned int u = __builtin_bit_cast(unsigned int, f);
  unsigned int r = u + 0x7fffu + ((u >> 16) & 1u);   // round-to-nearest-even
  return (unsigned short)(r >> 16);
}
DEV float bf2f(unsigned int lo16) {
  return __builtin_bit_cast(float, lo16 << 16);
}

// A-matrix 16x32 bf16 fragment (ISA 7.12.2): lane l: m=l&15, g=l>>4;
// reg r holds k = (r&3)*2 + g*8 + (r>>2)*16 and k+1.
DEV v16bf load_a_frag(const unsigned short* base, int stride) {
  int l = threadIdx.x & 31, m = l & 15, g = l >> 4;
  v8u a;
#pragma unroll
  for (int r = 0; r < 8; ++r) {
    int k = ((r & 3) << 1) + (g << 3) + ((r >> 2) << 4);
    a[r] = *(const unsigned int*)(base + m * stride + k);
  }
  return __builtin_bit_cast(v16bf, a);
}

// B-matrix 32x16 bf16 fragment: lane l: n=l&15, g=l>>4; reg r holds
// k = (r&3)*2 + g*8 + (r>>2)*16 and k+1. Memory: element(k,n) at base[n*stride + k].
DEV v16bf load_b_frag(const unsigned short* base, int stride) {
  int l = threadIdx.x & 31, n = l & 15, g = l >> 4;
  v8u b;
#pragma unroll
  for (int r = 0; r < 8; ++r)
    b[r] = *(const unsigned int*)(base + n * stride + ((r & 3) << 1) + (g << 3) + ((r >> 2) << 4));
  return __builtin_bit_cast(v16bf, b);
}

DEV v8f wmma_bf16(v16bf a, v16bf b, v8f c) {
  return __builtin_amdgcn_wmma_f32_16x16x32_bf16(
      false, a, false, b, (short)0, c, false, false);
}

// ---- gfx1250 async global->LDS copy (ASYNCcnt tracked) --------------------
// LDS operand = low 32 bits of the generic shared pointer.
DEV void async_copy16(unsigned short* ldst, const unsigned short* gsrc) {
  unsigned int l = (unsigned int)(unsigned long long)ldst;
  unsigned long long g = (unsigned long long)gsrc;
  asm volatile("global_load_async_to_lds_b128 %0, %1, off"
               :: "v"(l), "v"(g) : "memory");
}
DEV void wait_async_3() { asm volatile("s_wait_asynccnt 0x3" ::: "memory"); }
DEV void wait_async_0() { asm volatile("s_wait_asynccnt 0x0" ::: "memory"); }

// ---------------------------------------------------------------------------
// 1) f32 -> bf16 cast (2 elements / thread, packed 32-bit stores)
// ---------------------------------------------------------------------------
__global__ __launch_bounds__(256) void cast_f32_bf16(const float* __restrict__ x,
                                                     unsigned short* __restrict__ y,
                                                     int n) {
  int i = (blockIdx.x * blockDim.x + threadIdx.x) * 2;
  if (i < n) {
    unsigned int p = (unsigned int)f2bf(x[i]) |
                     ((unsigned int)f2bf(x[i + 1]) << 16);
    *(unsigned int*)(y + i) = p;
  }
}

// ---------------------------------------------------------------------------
// 2) GEMM: C[M,N] = A[M,K](bf16) * Wb[N,K](bf16)^T
//    block tile 64x128, 8 waves of 32x32 (2x2 WMMA frags), K-step 32,
//    double-buffered async global->LDS staging.
// ---------------------------------------------------------------------------
DEV void issue_tile(const unsigned short* A, const unsigned short* Wb,
                    unsigned short* Asl, unsigned short* Bsl,
                    int blkM, int blkN, int Kd, int kt, int t) {
  // A tile: 64 rows x 32 bf16 = 256 chunks of 16B, 1 per thread
  int arow = t >> 2, ach = t & 3;
  async_copy16(&Asl[arow * 32 + ach * 8],
               A + (size_t)(blkM * 64 + arow) * Kd + kt * 32 + ach * 8);
  // B tile: 128 rows x 32 bf16 = 512 chunks of 16B, 2 per thread
#pragma unroll
  for (int q = 0; q < 2; ++q) {
    int cid = t + q * 256;
    int brow = cid >> 2, bch = cid & 3;
    async_copy16(&Bsl[brow * 32 + bch * 8],
                 Wb + (size_t)(blkN * 128 + brow) * Kd + kt * 32 + bch * 8);
  }
}

template <bool OUT_F32>
__global__ __launch_bounds__(256) void gemm_bf16(const unsigned short* __restrict__ A,
                                                 const unsigned short* __restrict__ Wb,
                                                 void* __restrict__ Cout,
                                                 int N, int Kd) {
  __shared__ alignas(16) unsigned short As[2][64 * 32];
  __shared__ alignas(16) unsigned short Bs[2][128 * 32];

  const int t = threadIdx.x;
  const int blkN = blockIdx.x, blkM = blockIdx.y;
  const int wid = t >> 5;
  const int wm = (wid & 1) << 5;             // 0 / 32
  const int wn = (wid >> 1) << 5;            // 0/32/64/96

  v8f c00 = {}, c01 = {}, c10 = {}, c11 = {};

  const int nk = Kd >> 5;
  issue_tile(A, Wb, As[0], Bs[0], blkM, blkN, Kd, 0, t);

  for (int kt = 0; kt < nk; ++kt) {
    const int cur = kt & 1;
    if (kt + 1 < nk) {
      issue_tile(A, Wb, As[cur ^ 1], Bs[cur ^ 1], blkM, blkN, Kd, kt + 1, t);
      wait_async_3();                        // current tile's 3 ops done (in order)
    } else {
      wait_async_0();
    }
    __syncthreads();                         // all waves' copies landed

    v16bf a0 = load_a_frag(&As[cur][wm * 32], 32);
    v16bf a1 = load_a_frag(&As[cur][(wm + 16) * 32], 32);
    v16bf b0 = load_b_frag(&Bs[cur][wn * 32], 32);
    v16bf b1 = load_b_frag(&Bs[cur][(wn + 16) * 32], 32);
    c00 = wmma_bf16(a0, b0, c00);
    c01 = wmma_bf16(a0, b1, c01);
    c10 = wmma_bf16(a1, b0, c10);
    c11 = wmma_bf16(a1, b1, c11);
    __syncthreads();                         // safe to overwrite this buffer next iter
  }

  // store: C/D layout lane l: n=l&15, reg r -> m = r + 8*(l>>4)
  const int l = t & 31, cn = l & 15, g = l >> 4;
#pragma unroll
  for (int i = 0; i < 2; ++i)
#pragma unroll
    for (int j = 0; j < 2; ++j) {
      v8f cc = (i == 0) ? (j == 0 ? c00 : c01) : (j == 0 ? c10 : c11);
#pragma unroll
      for (int r = 0; r < 8; ++r) {
        int row = blkM * 64 + wm + i * 16 + r + (g << 3);
        int col = blkN * 128 + wn + j * 16 + cn;
        if (OUT_F32)
          ((float*)Cout)[(size_t)row * N + col] = cc[r];
        else
          ((unsigned short*)Cout)[(size_t)row * N + col] = f2bf(cc[r]);
      }
    }
}

// ---------------------------------------------------------------------------
// 3) RoPE in place on bf16 Q/K: pair (2i,2i+1) per thread, one 32-bit ld/st
// ---------------------------------------------------------------------------
__global__ __launch_bounds__(256) void rope_kernel(unsigned short* __restrict__ q,
                                                   int nheads,
                                                   const float* __restrict__ cosp,
                                                   const float* __restrict__ sinp,
                                                   int npairs) {
  int idx = blockIdx.x * blockDim.x + threadIdx.x;
  if (idx >= npairs) return;
  int i = idx & 63;                 // HD/2 = 64
  int rest = idx >> 6;
  int h = rest % nheads;
  int row = rest / nheads;          // b*S + s
  int s = row & (SS - 1);
  unsigned int* p = (unsigned int*)(q + (size_t)row * (nheads * HDD) + h * HDD + 2 * i);
  unsigned int v = *p;
  float xr = bf2f(v & 0xffffu), xi = bf2f(v >> 16);
  float c = cosp[s * 64 + i], sn = sinp[s * 64 + i];
  float o0 = xr * c - xi * sn;
  float o1 = xr * sn + xi * c;
  *p = (unsigned int)f2bf(o0) | ((unsigned int)f2bf(o1) << 16);
}

// ---------------------------------------------------------------------------
// 4) V transpose per batch: V[b*S+s, c] -> Vt[b*1024 + c, s]   (bf16)
// ---------------------------------------------------------------------------
__global__ __launch_bounds__(256) void transpose_v(const unsigned short* __restrict__ V,
                                                   unsigned short* __restrict__ Vt) {
  __shared__ unsigned short tile[32][33];
  int b = blockIdx.z;
  int c0 = blockIdx.x * 32, s0 = blockIdx.y * 32;
  int tx = threadIdx.x & 31, ty = threadIdx.x >> 5;
#pragma unroll
  for (int j = 0; j < 4; ++j) {
    int s = ty * 4 + j;
    tile[s][tx] = V[(size_t)(b * SS + s0 + s) * (NKVV * HDD) + c0 + tx];
  }
  __syncthreads();
#pragma unroll
  for (int j = 0; j < 4; ++j) {
    int c = ty * 4 + j;
    Vt[(size_t)(b * (NKVV * HDD) + c0 + c) * SS + s0 + tx] = tile[tx][c];
  }
}

// ---------------------------------------------------------------------------
// 5) Flash attention: 1 wave = 16 queries x HD=128, key blocks of 32.
//    grid = (S/64, NH, B), block = 128 (4 waves)
// ---------------------------------------------------------------------------
__global__ __launch_bounds__(128) void attn_kernel(const unsigned short* __restrict__ Q,
                                                   const unsigned short* __restrict__ K,
                                                   const unsigned short* __restrict__ Vt,
                                                   const unsigned char* __restrict__ mask,
                                                   unsigned short* __restrict__ O) {
  __shared__ alignas(16) unsigned short Plds[4][16 * 32];   // per-wave prob tile

  const int wid = threadIdx.x >> 5;
  const int l = threadIdx.x & 31, hn = l & 15, g = l >> 4;
  const int b = blockIdx.z, h = blockIdx.y;
  const int kvh = h >> 1;                                   // n_rep = 2
  const int q0 = blockIdx.x * 64 + wid * 16;

  const unsigned short* qbase = Q + (size_t)(b * SS + q0) * DIMM + h * HDD;
  v16bf qf[4];
#pragma unroll
  for (int c = 0; c < 4; ++c) qf[c] = load_a_frag(qbase + c * 32, DIMM);

  v8f acc[8];
#pragma unroll
  for (int c = 0; c < 8; ++c) acc[c] = v8f{};
  float m_i[8], l_i[8];
#pragma unroll
  for (int r = 0; r < 8; ++r) { m_i[r] = -1e30f; l_i[r] = 0.0f; }

  const unsigned char* mbase = mask + b * SS;
  unsigned short* pl = &Plds[wid][0];

  for (int kb = 0; kb < SS / 32; ++kb) {
    const int key0 = kb * 32;
    // ---- scores: S = Q * K^T (two 16-key tiles) ----
    v8f s0 = {}, s1 = {};
    const unsigned short* kbase = K + (size_t)(b * SS + key0) * (NKVV * HDD) + kvh * HDD;
#pragma unroll
    for (int c = 0; c < 4; ++c) {
      v16bf k0f = load_b_frag(kbase + c * 32, NKVV * HDD);
      v16bf k1f = load_b_frag(kbase + (size_t)16 * (NKVV * HDD) + c * 32, NKVV * HDD);
      s0 = wmma_bf16(qf[c], k0f, s0);
      s1 = wmma_bf16(qf[c], k1f, s1);
    }
    const bool msk0 = mbase[key0 + hn] != 0;
    const bool msk1 = mbase[key0 + 16 + hn] != 0;
#pragma unroll
    for (int r = 0; r < 8; ++r) {
      s0[r] = msk0 ? -1e30f : s0[r] * SCALE;
      s1[r] = msk1 ? -1e30f : s1[r] * SCALE;
    }
    // ---- online softmax (row = r + 8*g, reduce across 16-lane groups) ----
#pragma unroll
    for (int r = 0; r < 8; ++r) {
      float bm = fmaxf(s0[r], s1[r]);
#pragma unroll
      for (int off = 1; off < 16; off <<= 1) bm = fmaxf(bm, __shfl_xor(bm, off, 32));
      float mnew = fmaxf(m_i[r], bm);
      float corr = __expf(m_i[r] - mnew);
      float p0 = __expf(s0[r] - mnew);
      float p1 = __expf(s1[r] - mnew);
      float rs = p0 + p1;
#pragma unroll
      for (int off = 1; off < 16; off <<= 1) rs += __shfl_xor(rs, off, 32);
      l_i[r] = l_i[r] * corr + rs;
      m_i[r] = mnew;
      s0[r] = p0; s1[r] = p1;
#pragma unroll
      for (int c = 0; c < 8; ++c) acc[c][r] *= corr;
    }
    // ---- probs -> bf16, C-layout -> A-layout via per-wave LDS tile ----
#pragma unroll
    for (int r = 0; r < 8; ++r) {
      int m = r + (g << 3);
      pl[m * 32 + hn]      = f2bf(s0[r]);
      pl[m * 32 + 16 + hn] = f2bf(s1[r]);
    }
    v16bf pf = load_a_frag(pl, 32);
    // ---- out += P * V (B frags from transposed V, contiguous along keys) ----
    const unsigned short* vtb = Vt + (size_t)(b * (NKVV * HDD) + kvh * HDD) * SS + key0;
#pragma unroll
    for (int c = 0; c < 8; ++c) {
      v16bf vf = load_b_frag(vtb + (size_t)(c * 16) * SS, SS);
      acc[c] = wmma_bf16(pf, vf, acc[c]);
    }
  }

  // ---- normalize and store bf16 ----
  float rinv[8];
#pragma unroll
  for (int r = 0; r < 8; ++r) rinv[r] = 1.0f / l_i[r];
  unsigned short* obase = O + (size_t)(b * SS + q0) * DIMM + h * HDD;
#pragma unroll
  for (int c = 0; c < 8; ++c)
#pragma unroll
    for (int r = 0; r < 8; ++r) {
      int m = r + (g << 3);
      obase[(size_t)m * DIMM + c * 16 + hn] = f2bf(acc[c][r] * rinv[r]);
    }
}

// ---------------------------------------------------------------------------
// launcher
// ---------------------------------------------------------------------------
extern "C" void kernel_launch(void* const* d_in, const int* in_sizes, int n_in,
                              void* d_out, int out_size, void* d_ws, size_t ws_size,
                              hipStream_t stream) {
  const float* x    = (const float*)d_in[0];
  const float* wq   = (const float*)d_in[1];
  const float* wk   = (const float*)d_in[2];
  const float* wv   = (const float*)d_in[3];
  const float* wo   = (const float*)d_in[4];
  const float* fcos = (const float*)d_in[5];
  const float* fsin = (const float*)d_in[6];
  const unsigned char* mask = (const unsigned char*)d_in[7];
  float* out = (float*)d_out;

  char* ws = (char*)d_ws;
  unsigned short* xbf = (unsigned short*)(ws);                        // 16 MB
  unsigned short* Qw  = (unsigned short*)(ws + (size_t)16 * 1048576); // 16 MB
  unsigned short* Kw  = (unsigned short*)(ws + (size_t)32 * 1048576); //  8 MB
  unsigned short* Vw  = (unsigned short*)(ws + (size_t)40 * 1048576); //  8 MB
  unsigned short* Vt  = (unsigned short*)(ws + (size_t)48 * 1048576); //  8 MB
  unsigned short* Wb  = (unsigned short*)(ws + (size_t)56 * 1048576); //  8 MB, reused per weight
  unsigned short* AO  = xbf;  // reuse: x-bf16 dead after QKV GEMMs

  // 1) x -> bf16
  {
    int n = ROWS * DIMM;                      // 8,388,608
    cast_f32_bf16<<<n / 512, 256, 0, stream>>>(x, xbf, n);
  }
  // 2) QKV projections (each weight cast once to bf16, then async-staged GEMM)
  cast_f32_bf16<<<(DIMM * DIMM) / 512, 256, 0, stream>>>(wq, Wb, DIMM * DIMM);
  gemm_bf16<false><<<dim3(DIMM / 128, ROWS / 64), 256, 0, stream>>>(xbf, Wb, Qw, DIMM, DIMM);
  cast_f32_bf16<<<(NKVV * HDD * DIMM) / 512, 256, 0, stream>>>(wk, Wb, NKVV * HDD * DIMM);
  gemm_bf16<false><<<dim3(NKVV * HDD / 128, ROWS / 64), 256, 0, stream>>>(xbf, Wb, Kw, NKVV * HDD, DIMM);
  cast_f32_bf16<<<(NKVV * HDD * DIMM) / 512, 256, 0, stream>>>(wv, Wb, NKVV * HDD * DIMM);
  gemm_bf16<false><<<dim3(NKVV * HDD / 128, ROWS / 64), 256, 0, stream>>>(xbf, Wb, Vw, NKVV * HDD, DIMM);
  // 3) RoPE on Q and K
  {
    int nq = ROWS * NHH * (HDD / 2);          // 4,194,304
    rope_kernel<<<nq / 256, 256, 0, stream>>>(Qw, NHH, fcos, fsin, nq);
    int nk = ROWS * NKVV * (HDD / 2);         // 2,097,152
    rope_kernel<<<nk / 256, 256, 0, stream>>>(Kw, NKVV, fcos, fsin, nk);
  }
  // 4) V transpose
  transpose_v<<<dim3(NKVV * HDD / 32, SS / 32, BB), 256, 0, stream>>>(Vw, Vt);
  // 5) attention
  attn_kernel<<<dim3(SS / 64, NHH, BB), 128, 0, stream>>>(Qw, Kw, Vt, mask, AO);
  // 6) output projection (f32 out)
  cast_f32_bf16<<<(DIMM * DIMM) / 512, 256, 0, stream>>>(wo, Wb, DIMM * DIMM);
  gemm_bf16<true><<<dim3(DIMM / 128, ROWS / 64), 256, 0, stream>>>(AO, Wb, out, DIMM, DIMM);
}
